// TransformerBlock_90245852823706
// MI455X (gfx1250) — matrix operand order repaired and checked by reading, not compiled, not run on hardware
//
#include <hip/hip_runtime.h>

// ---------------------------------------------------------------------------
// CDNA5 / gfx1250 transformer block (MoE + GQA attention), compute-bound:
// ~250 GFLOP vs ~114MB bf16 weight traffic -> run all GEMMs through
// v_wmma_f32_16x16x32_bf16 with fp32 accumulation. GEMM uses a double-buffered
// LDS pipeline; tile staging uses global_load_async_to_lds_b128 (ASYNCcnt)
// when the toolchain exposes the builtin, else load+ds_store.
// Requires ~612MB workspace.
// ---------------------------------------------------------------------------

#ifndef __has_builtin
#define __has_builtin(x) 0
#endif

#if __has_builtin(__builtin_amdgcn_global_load_async_to_lds_b128)
#define ASYNC_LDS 1
#else
#define ASYNC_LDS 0
#endif

#if __has_builtin(__builtin_amdgcn_s_wait_asynccnt)
#define WAIT_ASYNC(n) __builtin_amdgcn_s_wait_asynccnt(n)
#else
#define WAIT_ASYNC(n) asm volatile("s_wait_asynccnt " #n ::: "memory")
#endif

#define AS1 __attribute__((address_space(1)))
#define AS3 __attribute__((address_space(3)))

typedef unsigned short u16;
typedef int v4i __attribute__((vector_size(16)));      // matches builtin param type
typedef __attribute__((ext_vector_type(16))) __bf16 v16bf;
typedef __attribute__((ext_vector_type(8)))  float  v8f;

#define B_   2
#define S_   1024
#define D_   1024
#define H_   16
#define KV_  4
#define DK_  64
#define E_   8
#define HID_ 2048
#define T_   (B_ * S_)
#define REP_ (H_ / KV_)

#define TM 128
#define TN 128
#define TK 32

__device__ __forceinline__ u16 f2bf(float f) {
    unsigned u = __float_as_uint(f);
    u += 0x7fffu + ((u >> 16) & 1u);     // round-to-nearest-even
    return (u16)(u >> 16);
}

// ---------------------------------------------------------------------------
// Batched GEMM:  C[z][M,N] = A[z][M,K] (row major, lda) x B[z][N,K] (row major,
// ldb; i.e. B-transposed layout).  Per-operand batch offset:
//   off = (z/div1)*s1 + ((z%div1)/div2)*s2
// Output fp32 or bf16.  M must be a multiple of 128 and K a multiple of 32
// (true for every call below); N is bounds-checked.
// ---------------------------------------------------------------------------
__global__ __launch_bounds__(256) void gemm_nt(
    const u16* __restrict__ A, const u16* __restrict__ B, void* __restrict__ C,
    int M, int N, int K, int lda, int ldb, int ldc,
    int aD1, int aD2, long long aS1, long long aS2,
    int bD1, int bD2, long long bS1, long long bS2,
    int cD1, int cD2, long long cS1, long long cS2,
    int cIsBf16)
{
    __shared__ u16 As[2][TM][TK];   // 16 KB (ping-pong)
    __shared__ u16 Bs[2][TN][TK];   // 16 KB
#if ASYNC_LDS
    __shared__ u16 Dump[16];        // sink for OOB async lanes
#endif

    const int z = blockIdx.z;
    const u16* Ab = A + (long long)(z / aD1) * aS1 + (long long)((z % aD1) / aD2) * aS2;
    const u16* Bb = B + (long long)(z / bD1) * bS1 + (long long)((z % bD1) / bD2) * bS2;
    const long long cOff = (long long)(z / cD1) * cS1 + (long long)((z % cD1) / cD2) * cS2;

    const int tid  = threadIdx.x;
    const int lane = tid & 31;
    const int wave = tid >> 5;
    const int wm   = wave >> 1;      // 0..3 : 32-row slab
    const int wn   = wave & 1;       // 0..1 : 64-col slab
    const int lr   = lane & 15;
    const int hi   = lane >> 4;      // half-wave select

    const int tileM = blockIdx.y * TM;
    const int tileN = blockIdx.x * TN;

    // Per-thread staging assignment: two 16B chunks of each tile.
    const int c0 = tid,       r0 = c0 >> 2, cc0 = (c0 & 3) * 8;
    const int c1 = tid + 256, r1 = c1 >> 2, cc1 = (c1 & 3) * 8;
    const u16* aP0 = Ab + (long long)(tileM + r0) * lda + cc0;
    const u16* aP1 = Ab + (long long)(tileM + r1) * lda + cc1;
    const bool bi0 = (tileN + r0) < N;
    const bool bi1 = (tileN + r1) < N;
    const u16* bP0 = Bb + (long long)(bi0 ? tileN + r0 : N - 1) * ldb + cc0;
    const u16* bP1 = Bb + (long long)(bi1 ? tileN + r1 : N - 1) * ldb + cc1;

    v8f acc[2][4];
    #pragma unroll
    for (int m = 0; m < 2; ++m)
        #pragma unroll
        for (int n = 0; n < 4; ++n) {
            v8f zv = {0.f,0.f,0.f,0.f,0.f,0.f,0.f,0.f};
            acc[m][n] = zv;
        }

    union Frag { uint4 q[2]; v16bf v; };

    auto stage = [&](int buf, int k0) {
#if ASYNC_LDS
        // 4 async ops per wave per stage -> exact ASYNCcnt bookkeeping.
        __builtin_amdgcn_global_load_async_to_lds_b128(
            (AS1 v4i*)(aP0 + k0), (AS3 v4i*)&As[buf][r0][cc0], 0, 0);
        __builtin_amdgcn_global_load_async_to_lds_b128(
            (AS1 v4i*)(aP1 + k0), (AS3 v4i*)&As[buf][r1][cc1], 0, 0);
        __builtin_amdgcn_global_load_async_to_lds_b128(
            (AS1 v4i*)(bP0 + k0),
            bi0 ? (AS3 v4i*)&Bs[buf][r0][cc0] : (AS3 v4i*)&Dump[0], 0, 0);
        __builtin_amdgcn_global_load_async_to_lds_b128(
            (AS1 v4i*)(bP1 + k0),
            bi1 ? (AS3 v4i*)&Bs[buf][r1][cc1] : (AS3 v4i*)&Dump[0], 0, 0);
#else
        *(uint4*)&As[buf][r0][cc0] = *(const uint4*)(aP0 + k0);
        *(uint4*)&As[buf][r1][cc1] = *(const uint4*)(aP1 + k0);
        uint4 v0 = make_uint4(0u,0u,0u,0u), v1 = make_uint4(0u,0u,0u,0u);
        if (bi0) v0 = *(const uint4*)(bP0 + k0);
        if (bi1) v1 = *(const uint4*)(bP1 + k0);
        *(uint4*)&Bs[buf][r0][cc0] = v0;
        *(uint4*)&Bs[buf][r1][cc1] = v1;
#endif
    };

#if ASYNC_LDS
    // OOB B rows are never written by async loads: zero them once per buffer.
    {
        uint4 zz = make_uint4(0u,0u,0u,0u);
        if (!bi0) { *(uint4*)&Bs[0][r0][cc0] = zz; *(uint4*)&Bs[1][r0][cc0] = zz; }
        if (!bi1) { *(uint4*)&Bs[0][r1][cc1] = zz; *(uint4*)&Bs[1][r1][cc1] = zz; }
    }
#endif

    stage(0, 0);
    const int nk = K / TK;
    for (int i = 0; i < nk; ++i) {
        const int buf = i & 1;
        if (i + 1 < nk) {
            stage(1 - buf, (i + 1) * TK);   // prefetch next tile into other buffer
#if ASYNC_LDS
            WAIT_ASYNC(4);                  // in-order: tile i's 4 ops complete
#endif
        } else {
#if ASYNC_LDS
            WAIT_ASYNC(0);
#endif
        }
        __syncthreads();

        // A fragments: lane holds row (wm*32 + m*16 + lr), K chunks
        // [kbase..kbase+7] and [kbase+16..kbase+23], kbase = hi*8
        Frag af[2];
        #pragma unroll
        for (int m = 0; m < 2; ++m) {
            const u16* p = &As[buf][wm * 32 + m * 16 + lr][hi * 8];
            af[m].q[0] = *(const uint4*)p;
            af[m].q[1] = *(const uint4*)(p + 16);
        }
        // B fragments: lane holds col N = (n*16 + lr), K chunks
        // [hi*8..hi*8+7] and [hi*8+16..hi*8+23]
        #pragma unroll
        for (int n = 0; n < 4; ++n) {
            Frag bf;
            const u16* p = &Bs[buf][wn * 64 + n * 16 + lr][hi * 8];
            bf.q[0] = *(const uint4*)p;
            bf.q[1] = *(const uint4*)(p + 16);
            #pragma unroll
            for (int m = 0; m < 2; ++m)
                acc[m][n] = __builtin_amdgcn_wmma_f32_16x16x32_bf16(
                    false, af[m].v, false, bf.v, (short)0, acc[m][n], false, false);
        }
        __syncthreads();   // all waves done with buf before it is restaged
    }

    // C layout: v8f element r -> row = m*16 + hi*8 + r, col = n*16 + lr
    #pragma unroll
    for (int m = 0; m < 2; ++m)
        #pragma unroll
        for (int n = 0; n < 4; ++n) {
            int col = tileN + wn * 64 + n * 16 + lr;
            if (col < N) {
                #pragma unroll
                for (int r = 0; r < 8; ++r) {
                    int row = tileM + wm * 32 + m * 16 + hi * 8 + r;
                    long long off = cOff + (long long)row * ldc + col;
                    float val = acc[m][n][r];
                    if (cIsBf16) ((u16*)C)[off] = f2bf(val);
                    else         ((float*)C)[off] = val;
                }
            }
        }
}

// fp32 W[K][N] -> bf16 Wt[N][K] (tiled LDS transpose; batched over z)
__global__ __launch_bounds__(256) void convt(
    const float* __restrict__ W, u16* __restrict__ Wt, int K, int N)
{
    __shared__ float t[32][33];
    long long base = (long long)blockIdx.z * (long long)K * N;
    int n0 = blockIdx.x * 32, k0 = blockIdx.y * 32;
    int tx = threadIdx.x & 31, ty = threadIdx.x >> 5;
    #pragma unroll
    for (int i = 0; i < 4; ++i) {
        int k = k0 + ty + i * 8, n = n0 + tx;
        float v = 0.f;
        if (k < K && n < N) v = W[base + (long long)k * N + n];
        t[ty + i * 8][tx] = v;
    }
    __syncthreads();
    #pragma unroll
    for (int i = 0; i < 4; ++i) {
        int n = n0 + ty + i * 8, k = k0 + tx;
        if (n < N && k < K)
            Wt[base + (long long)n * K + k] = f2bf(t[tx][ty + i * 8]);
    }
}

// RMSNorm row kernel: out_bf16 = g * x / sqrt(mean(x^2)+eps)
__global__ __launch_bounds__(256) void rmsnorm_k(
    const float* __restrict__ x, const float* __restrict__ g,
    u16* __restrict__ out, int D)
{
    __shared__ float red[256];
    long long row = blockIdx.x;
    const float* xr = x + row * D;
    int tid = threadIdx.x;
    float s = 0.f;
    for (int i = tid; i < D; i += 256) { float v = xr[i]; s += v * v; }
    red[tid] = s; __syncthreads();
    for (int st = 128; st > 0; st >>= 1) {
        if (tid < st) red[tid] += red[tid + st];
        __syncthreads();
    }
    float inv = rsqrtf(red[0] / (float)D + 1e-5f);
    u16* orow = out + row * D;
    for (int i = tid; i < D; i += 256) orow[i] = f2bf(xr[i] * inv * g[i]);
}

// RoPE on q,k (interleaved-pair convention) + pack to bf16; V packed transposed
__global__ __launch_bounds__(256) void rope_k(
    const float* __restrict__ qraw,  // [T, D]        (s-major, h*DK cols)
    const float* __restrict__ kvraw, // [T, 2*KV*DK]
    u16* __restrict__ qb,            // [B,H,S,DK]
    u16* __restrict__ kb,            // [B,KV,S,DK]
    u16* __restrict__ vtb,           // [B,KV,DK,S]   (transposed)
    const int* __restrict__ startp)
{
    int t = blockIdx.x;
    int b = t / S_, s = t % S_;
    float pos = (float)(s + startp[0]);
    int tid = threadIdx.x;
    const float lth = __logf(10000.0f);

    for (int p = tid; p < D_ / 2; p += 256) {
        int h = p / (DK_ / 2), pr = p % (DK_ / 2);
        float fr = __expf(-(2.0f * pr / (float)DK_) * lth);
        float sn, cs; __sincosf(pos * fr, &sn, &cs);
        float te = qraw[(long long)t * D_ + h * DK_ + 2 * pr];
        float to = qraw[(long long)t * D_ + h * DK_ + 2 * pr + 1];
        long long o = (((long long)(b * H_ + h)) * S_ + s) * DK_ + 2 * pr;
        qb[o]     = f2bf(te * cs - to * sn);
        qb[o + 1] = f2bf(te * sn + to * cs);
    }
    for (int p = tid; p < KV_ * DK_ / 2; p += 256) {
        int kv = p / (DK_ / 2), pr = p % (DK_ / 2);
        float fr = __expf(-(2.0f * pr / (float)DK_) * lth);
        float sn, cs; __sincosf(pos * fr, &sn, &cs);
        float te = kvraw[(long long)t * (2 * KV_ * DK_) + kv * DK_ + 2 * pr];
        float to = kvraw[(long long)t * (2 * KV_ * DK_) + kv * DK_ + 2 * pr + 1];
        long long o = (((long long)(b * KV_ + kv)) * S_ + s) * DK_ + 2 * pr;
        kb[o]     = f2bf(te * cs - to * sn);
        kb[o + 1] = f2bf(te * sn + to * cs);
    }
    for (int i = tid; i < KV_ * DK_; i += 256) {
        int kv = i / DK_, dk = i % DK_;
        float v = kvraw[(long long)t * (2 * KV_ * DK_) + KV_ * DK_ + kv * DK_ + dk];
        vtb[(((long long)(b * KV_ + kv)) * DK_ + dk) * S_ + s] = f2bf(v);
    }
}

// causal softmax over one score row; scale = 1/sqrt(DK) fused; bf16 probs out
__global__ __launch_bounds__(256) void softmax_causal(
    const float* __restrict__ scores, u16* __restrict__ probs)
{
    __shared__ float red[256];
    int s = blockIdx.x;
    long long base = ((long long)blockIdx.y * S_ + s) * S_;
    int tid = threadIdx.x;
    const float scale = 0.125f;

    float mx = -3.4e38f;
    for (int t = tid; t <= s; t += 256) mx = fmaxf(mx, scores[base + t] * scale);
    red[tid] = mx; __syncthreads();
    for (int st = 128; st > 0; st >>= 1) {
        if (tid < st) red[tid] = fmaxf(red[tid], red[tid + st]);
        __syncthreads();
    }
    mx = red[0]; __syncthreads();

    float sum = 0.f;
    for (int t = tid; t <= s; t += 256) sum += __expf(scores[base + t] * scale - mx);
    red[tid] = sum; __syncthreads();
    for (int st = 128; st > 0; st >>= 1) {
        if (tid < st) red[tid] += red[tid + st];
        __syncthreads();
    }
    float inv = 1.0f / red[0];

    for (int t = tid; t < S_; t += 256) {
        float p = (t <= s) ? __expf(scores[base + t] * scale - mx) * inv : 0.f;
        probs[base + t] = f2bf(p);
    }
}

__global__ void add2_k(const float* __restrict__ a, const float* __restrict__ b,
                       float* __restrict__ o, long long n)
{
    long long i = (long long)blockIdx.x * blockDim.x + threadIdx.x;
    if (i < n) o[i] = a[i] + b[i];
}

// router: softmax over E, write softmax probs (output #2), top-2 normalized -> cw
__global__ void router_k(const float* __restrict__ rl, float* __restrict__ lo,
                         float* __restrict__ cw)
{
    int t = blockIdx.x * blockDim.x + threadIdx.x;
    if (t >= T_) return;
    float v[E_];
    float mx = -3.4e38f;
    for (int e = 0; e < E_; ++e) { v[e] = rl[t * E_ + e]; mx = fmaxf(mx, v[e]); }
    float sm = 0.f;
    for (int e = 0; e < E_; ++e) { v[e] = __expf(v[e] - mx); sm += v[e]; }
    for (int e = 0; e < E_; ++e) { v[e] /= sm; lo[t * E_ + e] = v[e]; }
    int i1 = 0;
    for (int e = 1; e < E_; ++e) if (v[e] > v[i1]) i1 = e;
    int i2 = (i1 == 0) ? 1 : 0;
    for (int e = 0; e < E_; ++e) if (e != i1 && v[e] > v[i2]) i2 = e;
    float w = v[i1] + v[i2];
    for (int e = 0; e < E_; ++e) cw[t * E_ + e] = 0.f;
    cw[t * E_ + i1] = v[i1] / w;
    cw[t * E_ + i2] = v[i2] / w;
}

__global__ void silumul_k(const float* __restrict__ a, const float* __restrict__ b,
                          u16* __restrict__ o, long long n)
{
    long long i = (long long)blockIdx.x * blockDim.x + threadIdx.x;
    if (i < n) {
        float x = a[i];
        float s = x / (1.f + __expf(-x));
        o[i] = f2bf(s * b[i]);
    }
}

// out = h + shared + sum_e cw[t,e] * eo[e,t,d]
__global__ void combine_k(const float* __restrict__ h, const float* __restrict__ sh,
                          const float* __restrict__ eo, const float* __restrict__ cw,
                          float* __restrict__ out)
{
    long long i = (long long)blockIdx.x * blockDim.x + threadIdx.x;
    if (i >= (long long)T_ * D_) return;
    int t = (int)(i / D_);
    int d = (int)(i - (long long)t * D_);
    float r = 0.f;
    #pragma unroll
    for (int e = 0; e < E_; ++e)
        r += cw[t * E_ + e] * eo[((long long)e * T_ + t) * D_ + d];
    out[i] = h[i] + sh[i] + r;
}

// ---------------------------------------------------------------------------
extern "C" void kernel_launch(void* const* d_in, const int* in_sizes, int n_in,
                              void* d_out, int out_size, void* d_ws, size_t ws_size,
                              hipStream_t stream)
{
    (void)in_sizes; (void)n_in; (void)out_size; (void)ws_size;
    const float* x        = (const float*)d_in[0];
    const float* wq       = (const float*)d_in[1];
    const float* wkv      = (const float*)d_in[2];
    const float* wc       = (const float*)d_in[3];
    const float* g_attn   = (const float*)d_in[4];
    const float* g_ff     = (const float*)d_in[5];
    const float* router_w = (const float*)d_in[6];
    const float* sw1      = (const float*)d_in[7];
    const float* sw2      = (const float*)d_in[8];
    const float* sw3      = (const float*)d_in[9];
    const float* ew1      = (const float*)d_in[10];
    const float* ew2      = (const float*)d_in[11];
    const float* ew3      = (const float*)d_in[12];
    const int*   startp   = (const int*)d_in[13];

    char* ws = (char*)d_ws;
    size_t off = 0;
    auto alloc = [&](size_t bytes) -> char* {
        off = (off + 255) & ~(size_t)255;
        char* p = ws + off;
        off += bytes;
        return p;
    };

    u16* wqT   = (u16*)alloc((size_t)D_ * D_ * 2);
    u16* wkvT  = (u16*)alloc((size_t)(2 * KV_ * DK_) * D_ * 2);
    u16* wcT   = (u16*)alloc((size_t)D_ * D_ * 2);
    u16* rwT   = (u16*)alloc((size_t)E_ * D_ * 2);
    u16* sw1T  = (u16*)alloc((size_t)HID_ * D_ * 2);
    u16* sw2T  = (u16*)alloc((size_t)HID_ * D_ * 2);
    u16* sw3T  = (u16*)alloc((size_t)D_ * HID_ * 2);
    u16* ew1T  = (u16*)alloc((size_t)E_ * HID_ * D_ * 2);
    u16* ew2T  = (u16*)alloc((size_t)E_ * HID_ * D_ * 2);
    u16* ew3T  = (u16*)alloc((size_t)E_ * D_ * HID_ * 2);
    u16* hN    = (u16*)alloc((size_t)T_ * D_ * 2);
    float* qraw  = (float*)alloc((size_t)T_ * D_ * 4);
    float* kvraw = (float*)alloc((size_t)T_ * 2 * KV_ * DK_ * 4);
    u16* qb    = (u16*)alloc((size_t)B_ * H_ * S_ * DK_ * 2);
    u16* kb    = (u16*)alloc((size_t)B_ * KV_ * S_ * DK_ * 2);
    u16* vtb   = (u16*)alloc((size_t)B_ * KV_ * DK_ * S_ * 2);
    float* scores = (float*)alloc((size_t)B_ * H_ * S_ * S_ * 4);  // reused as ea
    u16*   probs  = (u16*)alloc((size_t)B_ * H_ * S_ * S_ * 2);    // reused as eh
    u16* attb    = (u16*)alloc((size_t)T_ * D_ * 2);
    float* attpj = (float*)alloc((size_t)T_ * D_ * 4);
    float* hbuf  = (float*)alloc((size_t)T_ * D_ * 4);
    u16* xn      = (u16*)alloc((size_t)T_ * D_ * 2);
    float* rl    = (float*)alloc((size_t)T_ * E_ * 4);
    float* cw    = (float*)alloc((size_t)T_ * E_ * 4);
    float* ga    = (float*)alloc((size_t)T_ * HID_ * 4);
    float* gb    = (float*)alloc((size_t)T_ * HID_ * 4);
    u16* h1s     = (u16*)alloc((size_t)T_ * HID_ * 2);
    float* shrd  = (float*)alloc((size_t)T_ * D_ * 4);
    float* eb    = (float*)alloc((size_t)E_ * T_ * HID_ * 4);
    float* eo    = (float*)alloc((size_t)E_ * T_ * D_ * 4);
    float* ea = scores;           // E*T*HID == B*H*S*S elements (exact match)
    u16*   eh = probs;            // ditto

    auto gemm = [&](const u16* A, const u16* Bt, void* C, int M, int N, int K,
                    int lda, int ldb, int ldc, int batch,
                    int aD1, int aD2, long long aS1, long long aS2,
                    int bD1, int bD2, long long bS1, long long bS2,
                    int cD1, int cD2, long long cS1, long long cS2, int cbf) {
        dim3 g((N + TN - 1) / TN, (M + TM - 1) / TM, batch);
        gemm_nt<<<g, 256, 0, stream>>>(A, Bt, C, M, N, K, lda, ldb, ldc,
                                       aD1, aD2, aS1, aS2, bD1, bD2, bS1, bS2,
                                       cD1, cD2, cS1, cS2, cbf);
    };

    // ---- weight convert+transpose to bf16 [N][K] ----
    convt<<<dim3(D_/32,   D_/32,   1 ), 256, 0, stream>>>(wq,       wqT,  D_,   D_);
    convt<<<dim3(512/32,  D_/32,   1 ), 256, 0, stream>>>(wkv,      wkvT, D_,   512);
    convt<<<dim3(D_/32,   D_/32,   1 ), 256, 0, stream>>>(wc,       wcT,  D_,   D_);
    convt<<<dim3(1,       D_/32,   1 ), 256, 0, stream>>>(router_w, rwT,  D_,   E_);
    convt<<<dim3(HID_/32, D_/32,   1 ), 256, 0, stream>>>(sw1,      sw1T, D_,   HID_);
    convt<<<dim3(HID_/32, D_/32,   1 ), 256, 0, stream>>>(sw2,      sw2T, D_,   HID_);
    convt<<<dim3(D_/32,   HID_/32, 1 ), 256, 0, stream>>>(sw3,      sw3T, HID_, D_);
    convt<<<dim3(HID_/32, D_/32,   E_), 256, 0, stream>>>(ew1,      ew1T, D_,   HID_);
    convt<<<dim3(HID_/32, D_/32,   E_), 256, 0, stream>>>(ew2,      ew2T, D_,   HID_);
    convt<<<dim3(D_/32,   HID_/32, E_), 256, 0, stream>>>(ew3,      ew3T, HID_, D_);

    // ---- attention ----
    rmsnorm_k<<<T_, 256, 0, stream>>>(x, g_attn, hN, D_);
    gemm(hN, wqT,  qraw,  T_, D_,  D_, D_, D_, D_,  1, 1,1,0,0, 1,1,0,0, 1,1,0,0, 0);
    gemm(hN, wkvT, kvraw, T_, 512, D_, D_, D_, 512, 1, 1,1,0,0, 1,1,0,0, 1,1,0,0, 0);
    rope_k<<<T_, 256, 0, stream>>>(qraw, kvraw, qb, kb, vtb, startp);

    // scores[z][s][t] = q[z] . k[gqa(z)],  z = b*H + h
    gemm(qb, kb, scores, S_, S_, DK_, DK_, DK_, S_, B_ * H_,
         B_ * H_, 1, 0, (long long)S_ * DK_,
         H_, REP_, (long long)KV_ * S_ * DK_, (long long)S_ * DK_,
         B_ * H_, 1, 0, (long long)S_ * S_, 0);
    softmax_causal<<<dim3(S_, B_ * H_), 256, 0, stream>>>(scores, probs);

    // att -> bf16, scattered directly into [B,S,H*DK] layout
    gemm(probs, vtb, attb, S_, DK_, S_, S_, S_, D_, B_ * H_,
         B_ * H_, 1, 0, (long long)S_ * S_,
         H_, REP_, (long long)KV_ * DK_ * S_, (long long)DK_ * S_,
         H_, 1, (long long)S_ * D_, (long long)DK_, 1);

    gemm(attb, wcT, attpj, T_, D_, D_, D_, D_, D_, 1, 1,1,0,0, 1,1,0,0, 1,1,0,0, 0);
    add2_k<<<(int)(((long long)T_ * D_ + 255) / 256), 256, 0, stream>>>(
        x, attpj, hbuf, (long long)T_ * D_);

    // ---- MoE ----
    rmsnorm_k<<<T_, 256, 0, stream>>>(hbuf, g_ff, xn, D_);
    gemm(xn, rwT, rl, T_, E_, D_, D_, D_, E_, 1, 1,1,0,0, 1,1,0,0, 1,1,0,0, 0);
    router_k<<<(T_ + 255) / 256, 256, 0, stream>>>(
        rl, (float*)d_out + (size_t)T_ * D_, cw);

    // shared expert
    gemm(xn, sw1T, ga, T_, HID_, D_, D_, D_, HID_, 1, 1,1,0,0, 1,1,0,0, 1,1,0,0, 0);
    gemm(xn, sw2T, gb, T_, HID_, D_, D_, D_, HID_, 1, 1,1,0,0, 1,1,0,0, 1,1,0,0, 0);
    silumul_k<<<(int)(((long long)T_ * HID_ + 255) / 256), 256, 0, stream>>>(
        ga, gb, h1s, (long long)T_ * HID_);
    gemm(h1s, sw3T, shrd, T_, D_, HID_, HID_, HID_, D_, 1,
         1,1,0,0, 1,1,0,0, 1,1,0,0, 0);

    // routed experts (dense over all E, like the reference)
    gemm(xn, ew1T, ea, T_, HID_, D_, D_, D_, HID_, E_,
         E_, 1, 0, 0,
         E_, 1, 0, (long long)HID_ * D_,
         E_, 1, 0, (long long)T_ * HID_, 0);
    gemm(xn, ew2T, eb, T_, HID_, D_, D_, D_, HID_, E_,
         E_, 1, 0, 0,
         E_, 1, 0, (long long)HID_ * D_,
         E_, 1, 0, (long long)T_ * HID_, 0);
    silumul_k<<<(int)(((long long)E_ * T_ * HID_ + 255) / 256), 256, 0, stream>>>(
        ea, eb, eh, (long long)E_ * T_ * HID_);
    gemm(eh, ew3T, eo, T_, D_, HID_, HID_, HID_, D_, E_,
         E_, 1, 0, (long long)T_ * HID_,
         E_, 1, 0, (long long)D_ * HID_,
         E_, 1, 0, (long long)T_ * D_, 0);

    combine_k<<<(int)(((long long)T_ * D_ + 255) / 256), 256, 0, stream>>>(
        hbuf, shrd, eo, cw, (float*)d_out);
}
